// DistTrainModel_65850438582672
// MI455X (gfx1250) — matrix operand order repaired and checked by reading, not compiled, not run on hardware
//
#include <hip/hip_runtime.h>
#include <math.h>

// ---------------------------------------------------------------------------
// DLRM forward (loss scalar) for MI455X / gfx1250.
// bf16 WMMA GEMMs + async global->LDS double-buffered staging (ASYNCcnt),
// k-loop unrolled x2 so buffer selection is compile-time (no constant-pool
// loads / cndmask on the ds_load address path).
// ---------------------------------------------------------------------------

typedef __attribute__((ext_vector_type(16))) __bf16 v16bf;
typedef __attribute__((ext_vector_type(8)))  float  v8f;

#define BATCH   4096
#define EMB_DIM 128
#define NSPARSE 26
#define TROWS   32      // 27 padded to 32 for WMMA
#define RDIM    512     // 479 padded to 512

// ---- WMMA fragment loaders (CDNA5 ISA 7.12.2 layouts, wave32) -------------

// A fragment: 16(M) x 32(K), row-major source, leading dim ld (in elements).
__device__ __forceinline__ v16bf load_frag_a(const __bf16* base, int ld, int lane) {
  v16bf a;
  const int m    = lane & 15;
  const int koff = (lane >> 4) << 3;       // 0 or 8
  const __bf16* row = base + m * ld;
#pragma unroll
  for (int j = 0; j < 8; ++j) {
    const int kb = ((j < 4) ? 0 : 16) + koff + ((j & 3) << 1);
    a[2 * j]     = row[kb];
    a[2 * j + 1] = row[kb + 1];
  }
  return a;
}

// B fragment: 32(K) x 16(N); source is W stored N-major: W[n][k], leading dim ld.
__device__ __forceinline__ v16bf load_frag_b(const __bf16* base, int ld, int lane) {
  v16bf b;
  const int n    = lane & 15;
  const int koff = (lane >> 4) << 3;       // 0 or 8
  const __bf16* row = base + n * ld;
#pragma unroll
  for (int j = 0; j < 8; ++j) {
    const int kb = ((j < 4) ? 0 : 16) + koff + ((j & 3) << 1);
    b[2 * j]     = row[kb];
    b[2 * j + 1] = row[kb + 1];
  }
  return b;
}

// ---------------------------------------------------------------------------
// Kernel 1: bottom layer 0 (K=13, too small for WMMA) : Y = relu(X W^T + b)
// ---------------------------------------------------------------------------
__global__ void dlrm_bot0(const float* __restrict__ X, const float* __restrict__ W,
                          const float* __restrict__ bias, __bf16* __restrict__ Y) {
  const int idx = blockIdx.x * blockDim.x + threadIdx.x;   // 4096*512 threads
  const int m = idx >> 9;
  const int n = idx & 511;
  float acc = bias[n];
  const float* xr = X + m * 13;
  const float* wr = W + n * 13;
#pragma unroll
  for (int k = 0; k < 13; ++k) acc += xr[k] * wr[k];
  Y[m * 512 + n] = (__bf16)fmaxf(acc, 0.0f);
}

// ---------------------------------------------------------------------------
// Kernel 2: weight f32 -> bf16 with optional K padding (zeros)
// ---------------------------------------------------------------------------
__global__ void dlrm_cvt_w(const float* __restrict__ src, __bf16* __restrict__ dst,
                           int N, int Kin, int Kpad) {
  const int idx = blockIdx.x * blockDim.x + threadIdx.x;
  if (idx >= N * Kpad) return;
  const int n = idx / Kpad;
  const int k = idx - n * Kpad;
  dst[idx] = (k < Kin) ? (__bf16)src[n * Kin + k] : (__bf16)0.0f;
}

// ---------------------------------------------------------------------------
// Kernel 3: bf16 WMMA GEMM  Y = act(A W^T + bias), async double-buffered LDS.
//   A: M x K bf16 row-major; W: N x K bf16 row-major; Y: bf16, ld = ldY
//   block tile 128(M) x 64(N), 8 waves, wave tile 32x32 (2x2 WMMA accums)
//   Requires M%128==0, N%64==0, K%64==0, 16B-aligned A/W.
// ---------------------------------------------------------------------------
#define AS_LD 40           // skewed LDS leading dim (elements); 80B rows, 16B-aligned
#define A_ELEMS (128 * AS_LD)
#define B_ELEMS (64 * AS_LD)
#define AE0 0u
#define AE1 ((unsigned)A_ELEMS)
#define BE0 ((unsigned)(2 * A_ELEMS))
#define BE1 ((unsigned)(2 * A_ELEMS + B_ELEMS))
#define GEMM_LDS_BYTES ((2 * A_ELEMS + 2 * B_ELEMS) * 2)

__global__ void dlrm_gemm_bf16(const __bf16* __restrict__ A, const __bf16* __restrict__ W,
                               const float* __restrict__ bias, __bf16* __restrict__ Y,
                               int M, int N, int K, int ldY, int relu) {
  extern __shared__ __bf16 smem[];   // [A0 | A1 | B0 | B1]

  const int tid  = threadIdx.x;
  const int lane = tid & 31;
  const int wave = tid >> 5;          // 0..7
  const int wm   = wave >> 1;         // 0..3 -> M offset 32*wm
  const int wn   = wave & 1;          // 0..1 -> N offset 32*wn
  const int bm   = blockIdx.y * 128;
  const int bn   = blockIdx.x * 64;

  // staging coordinates: A tile 128x32 (16 elems/thread), B tile 64x32 (8/thread)
  const int ar = tid >> 1, ac = (tid & 1) << 4;
  const int br = tid >> 2, bc = (tid & 3) << 3;
  const __bf16* gA = A + (size_t)(bm + ar) * K + ac;
  const __bf16* gB = W + (size_t)(bn + br) * K + bc;

  const unsigned ldsBase = __builtin_amdgcn_groupstaticsize();
  const unsigned slotA = (unsigned)(ar * AS_LD + ac) * 2u;
  const unsigned slotB = (unsigned)(br * AS_LD + bc) * 2u;
  const unsigned lA0 = ldsBase + AE0 * 2u + slotA;
  const unsigned lA1 = ldsBase + AE1 * 2u + slotA;
  const unsigned lB0 = ldsBase + BE0 * 2u + slotB;
  const unsigned lB1 = ldsBase + BE1 * 2u + slotB;

  // async stage of one 32-wide K slice (3x b128 per thread).
  // NOTE: inst offset applies to BOTH LDS and global addresses (ISA 10.x).
  auto stage = [&](unsigned la, unsigned lb, int k0) {
    const __bf16* pa = gA + k0;
    const __bf16* pb = gB + k0;
    asm volatile("global_load_async_to_lds_b128 %0, %1, off\n\t"
                 "global_load_async_to_lds_b128 %0, %1, off offset:16"
                 :: "v"(la), "v"(pa) : "memory");
    asm volatile("global_load_async_to_lds_b128 %0, %1, off"
                 :: "v"(lb), "v"(pb) : "memory");
  };

  v8f c00 = {}, c01 = {}, c10 = {}, c11 = {};

  auto compute = [&](unsigned aoff, unsigned boff) {
    const __bf16* Ab = smem + aoff;
    const __bf16* Bb = smem + boff;
    const v16bf a0 = load_frag_a(Ab + (32 * wm)      * AS_LD, AS_LD, lane);
    const v16bf a1 = load_frag_a(Ab + (32 * wm + 16) * AS_LD, AS_LD, lane);
    const v16bf b0 = load_frag_b(Bb + (32 * wn)      * AS_LD, AS_LD, lane);
    const v16bf b1 = load_frag_b(Bb + (32 * wn + 16) * AS_LD, AS_LD, lane);
    c00 = __builtin_amdgcn_wmma_f32_16x16x32_bf16(false, a0, false, b0, (short)0, c00, false, false);
    c01 = __builtin_amdgcn_wmma_f32_16x16x32_bf16(false, a0, false, b1, (short)0, c01, false, false);
    c10 = __builtin_amdgcn_wmma_f32_16x16x32_bf16(false, a1, false, b0, (short)0, c10, false, false);
    c11 = __builtin_amdgcn_wmma_f32_16x16x32_bf16(false, a1, false, b1, (short)0, c11, false, false);
  };

  stage(lA0, lB0, 0);
  for (int k0 = 0; k0 < K; k0 += 64) {     // K % 64 == 0; compile-time buffers
    asm volatile("s_wait_asynccnt 0x0" ::: "memory");
    __syncthreads();                       // all waves' async writes visible;
                                           // barrier lowering fences dscnt too
    stage(lA1, lB1, k0 + 32);              // K%64==0 -> k0+32 always valid
    compute(AE0, BE0);

    asm volatile("s_wait_asynccnt 0x0" ::: "memory");
    __syncthreads();
    if (k0 + 64 < K) stage(lA0, lB0, k0 + 64);
    compute(AE1, BE1);
  }

  // epilogue: bias + (relu) + bf16 store.  C layout: VGPR r -> M=r+8*(lane>>4), N=lane&15
  const int mo = bm + 32 * wm + ((lane >> 4) << 3);
  const int no = bn + 32 * wn + (lane & 15);
  const float bias0 = bias[no];
  const float bias1 = bias[no + 16];
#pragma unroll
  for (int r = 0; r < 8; ++r) {
    float v00 = c00[r] + bias0;
    float v01 = c01[r] + bias1;
    float v10 = c10[r] + bias0;
    float v11 = c11[r] + bias1;
    if (relu) {
      v00 = fmaxf(v00, 0.0f); v01 = fmaxf(v01, 0.0f);
      v10 = fmaxf(v10, 0.0f); v11 = fmaxf(v11, 0.0f);
    }
    Y[(size_t)(mo + r)      * ldY + no]      = (__bf16)v00;
    Y[(size_t)(mo + r)      * ldY + no + 16] = (__bf16)v01;
    Y[(size_t)(mo + r + 16) * ldY + no]      = (__bf16)v10;
    Y[(size_t)(mo + r + 16) * ldY + no + 16] = (__bf16)v11;
  }
}

// ---------------------------------------------------------------------------
// Kernel 4: embedding gather -> T rows 1..26  (bf16)
// ---------------------------------------------------------------------------
__global__ void dlrm_gather(const float* __restrict__ emb, const int* __restrict__ lS_i,
                            __bf16* __restrict__ T) {
  const int blk = blockIdx.x;                 // 26*4096
  const int s = blk >> 12;
  const int b = blk & 4095;
  const long long row = (long long)lS_i[s * BATCH + b];
  const float* src = emb + ((size_t)s * 100000 + (size_t)row) * EMB_DIM;
  __bf16* dst = T + ((size_t)b * TROWS + 1 + s) * EMB_DIM;
  dst[threadIdx.x] = (__bf16)src[threadIdx.x];   // 128 threads
}

// Kernel 5: zero T pad rows 27..31
__global__ void dlrm_tpad(__bf16* __restrict__ T) {
  const int idx = blockIdx.x * blockDim.x + threadIdx.x;     // 4096*5*128
  if (idx >= BATCH * 5 * EMB_DIM) return;
  const int b   = idx / (5 * EMB_DIM);
  const int rem = idx - b * (5 * EMB_DIM);
  const int row = 27 + rem / EMB_DIM;
  const int d   = rem - (rem / EMB_DIM) * EMB_DIM;
  T[((size_t)b * TROWS + row) * EMB_DIM + d] = (__bf16)0.0f;
}

// ---------------------------------------------------------------------------
// Kernel 6: interaction. One wave per batch: Z = T_b T_b^T via 32x32 WMMA tile
// (rows 27..31 are zero).  Scatter strict lower triangle into R[:,128:479].
// ---------------------------------------------------------------------------
__global__ void dlrm_interact(const __bf16* __restrict__ T, __bf16* __restrict__ R) {
  const int lane = threadIdx.x & 31;
  const int b    = (blockIdx.x * blockDim.x + threadIdx.x) >> 5;  // wave id == batch
  const __bf16* Tb = T + (size_t)b * TROWS * EMB_DIM;

  v8f c00 = {}, c01 = {}, c10 = {}, c11 = {};
#pragma unroll
  for (int k0 = 0; k0 < EMB_DIM; k0 += 32) {
    const v16bf a0 = load_frag_a(Tb + k0,                 EMB_DIM, lane);
    const v16bf a1 = load_frag_a(Tb + 16 * EMB_DIM + k0,  EMB_DIM, lane);
    const v16bf b0 = load_frag_b(Tb + k0,                 EMB_DIM, lane);
    const v16bf b1 = load_frag_b(Tb + 16 * EMB_DIM + k0,  EMB_DIM, lane);
    c00 = __builtin_amdgcn_wmma_f32_16x16x32_bf16(false, a0, false, b0, (short)0, c00, false, false);
    c01 = __builtin_amdgcn_wmma_f32_16x16x32_bf16(false, a0, false, b1, (short)0, c01, false, false);
    c10 = __builtin_amdgcn_wmma_f32_16x16x32_bf16(false, a1, false, b0, (short)0, c10, false, false);
    c11 = __builtin_amdgcn_wmma_f32_16x16x32_bf16(false, a1, false, b1, (short)0, c11, false, false);
  }

  __bf16* Rb = R + (size_t)b * RDIM + EMB_DIM;    // Zflat region (351 values)
  const int mbase = ((lane >> 4) << 3);
  const int nbase = (lane & 15);
#pragma unroll
  for (int r = 0; r < 8; ++r) {
    const int m0 = mbase + r;
    {
      const int m = m0, n = nbase;                      // quadrant (0,0)
      if (m > n && m < 27) Rb[(m * (m - 1)) / 2 + n] = (__bf16)c00[r];
    }
    // quadrant (0,1): m<16<=n -> never strictly-lower
    {
      const int m = m0 + 16, n = nbase;                 // quadrant (1,0)
      if (m > n && m < 27) Rb[(m * (m - 1)) / 2 + n] = (__bf16)c10[r];
    }
    {
      const int m = m0 + 16, n = nbase + 16;            // quadrant (1,1)
      if (m > n && m < 27) Rb[(m * (m - 1)) / 2 + n] = (__bf16)c11[r];
    }
    (void)c01;
  }
}

// Kernel 7: R[:, :128] = x (T row 0),  R[:, 479:512] = 0
__global__ void dlrm_prepR(const __bf16* __restrict__ T, __bf16* __restrict__ R) {
  const int b = blockIdx.x;
  const int t = threadIdx.x;            // 256
  __bf16* Rb = R + (size_t)b * RDIM;
  if (t < 128)      Rb[t] = T[(size_t)b * TROWS * EMB_DIM + t];
  else if (t < 161) Rb[479 + (t - 128)] = (__bf16)0.0f;
}

// ---------------------------------------------------------------------------
// Kernel 8: final layer (256 -> 1) + sigmoid + clipped BCE, atomic reduce.
// ---------------------------------------------------------------------------
__global__ void dlrm_loss(const __bf16* __restrict__ A4, const float* __restrict__ w4,
                          const float* __restrict__ b4, const float* __restrict__ target,
                          float* __restrict__ out) {
  const int lane = threadIdx.x & 31;
  const int b    = (blockIdx.x * blockDim.x + threadIdx.x) >> 5;
  const __bf16* row = A4 + (size_t)b * 256;
  float acc = 0.0f;
#pragma unroll
  for (int i = 0; i < 8; ++i) {
    const int k = lane + 32 * i;
    acc += (float)row[k] * w4[k];
  }
#pragma unroll
  for (int off = 16; off > 0; off >>= 1) acc += __shfl_down(acc, off, 32);
  if (lane == 0) {
    const float z = acc + b4[0];
    const float p = 1.0f / (1.0f + expf(-z));
    const float t = target[b];
    const float lp  = fmaxf(logf(p),     -100.0f);
    const float l1p = fmaxf(log1pf(-p),  -100.0f);
    atomicAdd(out, -(t * lp + (1.0f - t) * l1p) * (1.0f / (float)BATCH));
  }
}

__global__ void dlrm_zero_out(float* out) { out[0] = 0.0f; }

// ---------------------------------------------------------------------------
extern "C" void kernel_launch(void* const* d_in, const int* in_sizes, int n_in,
                              void* d_out, int out_size, void* d_ws, size_t ws_size,
                              hipStream_t stream) {
  const float* dense_x = (const float*)d_in[0];
  const int*   lS_i    = (const int*)  d_in[1];
  const float* target  = (const float*)d_in[2];
  const float* emb     = (const float*)d_in[3];
  const float* bw0 = (const float*)d_in[4];  const float* bb0 = (const float*)d_in[5];
  const float* bw1 = (const float*)d_in[6];  const float* bb1 = (const float*)d_in[7];
  const float* bw2 = (const float*)d_in[8];  const float* bb2 = (const float*)d_in[9];
  const float* tw0 = (const float*)d_in[10]; const float* tb0 = (const float*)d_in[11];
  const float* tw1 = (const float*)d_in[12]; const float* tb1 = (const float*)d_in[13];
  const float* tw2 = (const float*)d_in[14]; const float* tb2 = (const float*)d_in[15];
  const float* tw3 = (const float*)d_in[16]; const float* tb3 = (const float*)d_in[17];
  const float* tw4 = (const float*)d_in[18]; const float* tb4 = (const float*)d_in[19];
  float* out = (float*)d_out;

  // ---- workspace carve-up (bf16 element counts, 256B aligned) ----
  char* ws = (char*)d_ws;
  size_t off = 0;
  auto carve = [&](size_t elems) -> __bf16* {
    __bf16* p = (__bf16*)(ws + off);
    off = (off + elems * sizeof(__bf16) + 255) & ~(size_t)255;
    return p;
  };
  __bf16* Wb1 = carve(256 * 512);
  __bf16* Wb2 = carve(128 * 256);
  __bf16* Wt0 = carve(1024 * 512);
  __bf16* Wt1 = carve(1024 * 1024);
  __bf16* Wt2 = carve(512 * 1024);
  __bf16* Wt3 = carve(256 * 512);
  __bf16* Y0  = carve((size_t)BATCH * 512);
  __bf16* Y1  = carve((size_t)BATCH * 256);
  __bf16* T   = carve((size_t)BATCH * TROWS * EMB_DIM);
  __bf16* R   = carve((size_t)BATCH * RDIM);
  __bf16* A1  = carve((size_t)BATCH * 1024);
  __bf16* A2  = carve((size_t)BATCH * 1024);
  __bf16* A3  = carve((size_t)BATCH * 512);
  __bf16* A4  = carve((size_t)BATCH * 256);

  // ---- weight conversion (bf16, tw0 K-padded 479->512) ----
  auto cvt = [&](const float* s, __bf16* d, int N, int Kin, int Kpad) {
    int tot = N * Kpad;
    dlrm_cvt_w<<<(tot + 255) / 256, 256, 0, stream>>>(s, d, N, Kin, Kpad);
  };
  cvt(bw1, Wb1, 256, 512, 512);
  cvt(bw2, Wb2, 128, 256, 256);
  cvt(tw0, Wt0, 1024, 479, 512);
  cvt(tw1, Wt1, 1024, 1024, 1024);
  cvt(tw2, Wt2, 512, 1024, 1024);
  cvt(tw3, Wt3, 256, 512, 512);

  // ---- bottom MLP ----
  dlrm_bot0<<<(BATCH * 512) / 256, 256, 0, stream>>>(dense_x, bw0, bb0, Y0);
  dlrm_gemm_bf16<<<dim3(256 / 64, BATCH / 128), 256, GEMM_LDS_BYTES, stream>>>(
      Y0, Wb1, bb1, Y1, BATCH, 256, 512, 256, 1);
  // layer 2 writes directly into T row 0 (ldY = 32*128)
  dlrm_gemm_bf16<<<dim3(128 / 64, BATCH / 128), 256, GEMM_LDS_BYTES, stream>>>(
      Y1, Wb2, bb2, T, BATCH, 128, 256, TROWS * EMB_DIM, 1);

  // ---- embeddings + padding ----
  dlrm_gather<<<NSPARSE * BATCH, EMB_DIM, 0, stream>>>(emb, lS_i, T);
  dlrm_tpad<<<(BATCH * 5 * EMB_DIM + 255) / 256, 256, 0, stream>>>(T);

  // ---- interactions + feature assembly ----
  dlrm_interact<<<BATCH / 8, 256, 0, stream>>>(T, R);
  dlrm_prepR<<<BATCH, 256, 0, stream>>>(T, R);

  // ---- top MLP ----
  dlrm_gemm_bf16<<<dim3(1024 / 64, BATCH / 128), 256, GEMM_LDS_BYTES, stream>>>(
      R,  Wt0, tb0, A1, BATCH, 1024, 512, 1024, 1);
  dlrm_gemm_bf16<<<dim3(1024 / 64, BATCH / 128), 256, GEMM_LDS_BYTES, stream>>>(
      A1, Wt1, tb1, A2, BATCH, 1024, 1024, 1024, 1);
  dlrm_gemm_bf16<<<dim3(512 / 64, BATCH / 128), 256, GEMM_LDS_BYTES, stream>>>(
      A2, Wt2, tb2, A3, BATCH, 512, 1024, 512, 1);
  dlrm_gemm_bf16<<<dim3(256 / 64, BATCH / 128), 256, GEMM_LDS_BYTES, stream>>>(
      A3, Wt3, tb3, A4, BATCH, 256, 512, 256, 1);

  // ---- final dot + sigmoid + BCE ----
  dlrm_zero_out<<<1, 1, 0, stream>>>(out);
  dlrm_loss<<<BATCH / 8, 256, 0, stream>>>(A4, tw4, tb4, target, out);
}
